// Transformer_47717086658566
// MI455X (gfx1250) — matrix operand order repaired and checked by reading, not compiled, not run on hardware
//
#include <hip/hip_runtime.h>
#include <hip/hip_bf16.h>
#include <math.h>

// ---------------- types for CDNA5 WMMA ----------------
typedef __attribute__((ext_vector_type(16))) __bf16 v16bf;
typedef __attribute__((ext_vector_type(8)))  __bf16 v8bf;
typedef __attribute__((ext_vector_type(8)))  float  v8f;

union Frag { v16bf v; v8bf h[2]; };

static __device__ __forceinline__ v8f wmma_bf16(const Frag& a, const Frag& b, v8f c) {
    return __builtin_amdgcn_wmma_f32_16x16x32_bf16(false, a.v, false, b.v,
                                                   (short)0, c, false, false);
}

// async global->LDS copy of 16 bytes per lane (CDNA5 ASYNCcnt path).
// lds byte offset = low 32 bits of the flat pointer (LDS aperture mapping).
static __device__ __forceinline__ void async_cp16(const __bf16* gptr, __bf16* lptr) {
    unsigned loff = (unsigned)(unsigned long long)lptr;
    asm volatile("global_load_async_to_lds_b128 %0, %1, off"
                 :: "v"(loff), "v"(gptr) : "memory");
}
static __device__ __forceinline__ void async_wait0() {
    asm volatile("s_wait_asynccnt 0x0" ::: "memory");
}

// ---------------- weight conversion kernels ----------------
// Wq/Wk/Wv are [H=16][D=1024][HD=64]; produce Wt[n][d] bf16 with n = h*64+kk
__global__ __launch_bounds__(256) void qkv_cvt(const float* __restrict__ in,
                                               __bf16* __restrict__ out) {
    int i = blockIdx.x * 256 + threadIdx.x;      // over 1024*1024, d fast
    int d = i & 1023, n = i >> 10;
    out[i] = (__bf16)in[((n >> 6) << 16) + (d << 6) + (n & 63)];
}

// in f32 row-major [K][N]  ->  out bf16 [N][K] (transposed)
__global__ __launch_bounds__(256) void transpose_cvt(const float* __restrict__ in,
                                                     __bf16* __restrict__ out,
                                                     int K, int N) {
    size_t i = (size_t)blockIdx.x * 256 + threadIdx.x;   // k fast in output
    if (i >= (size_t)K * N) return;
    int k = (int)(i % K), n = (int)(i / K);
    out[i] = (__bf16)in[(size_t)k * N + n];
}

// ---------------- LayerNorm (f32 in -> bf16 out), D = 1024 ----------------
__global__ __launch_bounds__(256) void ln_bf16(const float* __restrict__ x,
                                               const float* __restrict__ g,
                                               const float* __restrict__ be,
                                               __bf16* __restrict__ out) {
    __shared__ float red[16];
    const int row = blockIdx.x, tid = threadIdx.x;
    const float* xr = x + (size_t)row * 1024;
    float v[4], s = 0.f, s2 = 0.f;
#pragma unroll
    for (int i = 0; i < 4; i++) { float t = xr[tid + 256 * i]; v[i] = t; s += t; s2 += t * t; }
#pragma unroll
    for (int d = 1; d < 32; d <<= 1) { s += __shfl_xor(s, d, 32); s2 += __shfl_xor(s2, d, 32); }
    const int wid = tid >> 5, lane = tid & 31;
    if (lane == 0) { red[wid] = s; red[8 + wid] = s2; }
    __syncthreads();
    if (wid == 0) {
        float a  = (lane < 8) ? red[lane] : 0.f;
        float b2 = (lane < 8) ? red[8 + lane] : 0.f;
#pragma unroll
        for (int d = 1; d < 8; d <<= 1) { a += __shfl_xor(a, d, 32); b2 += __shfl_xor(b2, d, 32); }
        if (lane == 0) { red[0] = a; red[1] = b2; }
    }
    __syncthreads();
    float mean = red[0] * (1.0f / 1024.0f);
    float var  = red[1] * (1.0f / 1024.0f) - mean * mean;
    float rstd = rsqrtf(var + 1e-5f);
#pragma unroll
    for (int i = 0; i < 4; i++) {
        int c = tid + 256 * i;
        out[(size_t)row * 1024 + c] = (__bf16)((v[i] - mean) * rstd * g[c] + be[c]);
    }
}

// ---------------- bf16 WMMA GEMM:  C[M,N] = A[M,K] * Bt[N,K]^T ----------------
// Double-buffered LDS, async global->LDS staging (ASYNCcnt), one barrier per K-step.
// epilogue modes:
//   0: out_bf[row*N+col] = bf16( scale*(acc + bias[col]) )
//   1: V transposed store: out_bf[((row>>11)*1024 + col)*2048 + (row&2047)] = bf16(acc+bias)
//   2: out_f[row*N+col] = acc + bias[col] + resid[row*N+col]
//   3: out_f[row*N+col] = gelu(acc + bias[col]) + resid[row*N+col]   (exact erf gelu)
__global__ __launch_bounds__(256) void gemm_bf16(const __bf16* __restrict__ A,
                                                 const __bf16* __restrict__ Bt,
                                                 const float* __restrict__ bias,
                                                 const float* __restrict__ resid,
                                                 float scale, int M, int N, int K,
                                                 int mode,
                                                 __bf16* __restrict__ outb,
                                                 float* __restrict__ outf) {
    __shared__ __align__(16) __bf16 As[2][128][40];
    __shared__ __align__(16) __bf16 Bs[2][128][40];
    const int tid = threadIdx.x;
    const int lane = tid & 31, wid = tid >> 5;
    const int wm = wid >> 2, wn = wid & 3;          // 2 x 4 wave grid
    const int lhi = lane >> 4, llo = lane & 15;
    const int m0 = blockIdx.y * 128, n0 = blockIdx.x * 128;
    const int srow = tid >> 2, scol = (tid & 3) << 3;   // staging chunk coords
    v8f acc[4][2] = {};

    auto stage = [&](int k0, int buf) {
#pragma unroll
        for (int half = 0; half < 2; half++) {
            int row = srow + half * 64;
            async_cp16(&A [(size_t)(m0 + row) * K + k0 + scol], &As[buf][row][scol]);
            async_cp16(&Bt[(size_t)(n0 + row) * K + k0 + scol], &Bs[buf][row][scol]);
        }
    };

    const int nk = K >> 5;
    stage(0, 0);
    for (int i = 0; i < nk; i++) {
        async_wait0();        // my tile-i copies landed in LDS
        __syncthreads();      // everyone's landed; everyone done with buf being refilled
        if (i + 1 < nk) stage((i + 1) << 5, (i + 1) & 1);
        const int buf = i & 1;

        Frag b[2];
#pragma unroll
        for (int nt = 0; nt < 2; nt++) {
            int n = wn * 32 + nt * 16 + llo;
            b[nt].h[0] = *(const v8bf*)&Bs[buf][n][8 * lhi];
            b[nt].h[1] = *(const v8bf*)&Bs[buf][n][16 + 8 * lhi];
        }
#pragma unroll
        for (int mt = 0; mt < 4; mt++) {
            int m = wm * 64 + mt * 16 + llo;
            Frag a;
            a.h[0] = *(const v8bf*)&As[buf][m][8 * lhi];
            a.h[1] = *(const v8bf*)&As[buf][m][16 + 8 * lhi];
#pragma unroll
            for (int nt = 0; nt < 2; nt++)
                acc[mt][nt] = wmma_bf16(a, b[nt], acc[mt][nt]);
        }
    }

#pragma unroll
    for (int mt = 0; mt < 4; mt++)
#pragma unroll
        for (int nt = 0; nt < 2; nt++)
#pragma unroll
            for (int r = 0; r < 8; r++) {
                int row = m0 + wm * 64 + mt * 16 + r + 8 * lhi;
                int col = n0 + wn * 32 + nt * 16 + llo;
                float v = acc[mt][nt][r];
                if (mode == 0) {
                    outb[(size_t)row * N + col] = (__bf16)(scale * (v + bias[col]));
                } else if (mode == 1) {
                    int bb = row >> 11, s = row & 2047;
                    outb[((size_t)(bb << 10) + col) * 2048 + s] = (__bf16)(v + bias[col]);
                } else if (mode == 2) {
                    size_t idx = (size_t)row * N + col;
                    outf[idx] = v + bias[col] + resid[idx];
                } else {
                    size_t idx = (size_t)row * N + col;
                    float t = v + bias[col];
                    outf[idx] = 0.5f * t * (1.0f + erff(t * 0.70710678118654752f)) + resid[idx];
                }
            }
}

// ---------------- flash attention ----------------
// Q,K: bf16 [B*S=8192][1024] (head h occupies cols h*64..h*64+63; 1/sqrt(HD) folded into Q)
// Vt:  bf16 [b][h][d=64][t=2048]
// O:   bf16 [8192][1024]  (concat heads, row-major)
// Double-buffered K/V tiles via async global->LDS, one barrier per t-step.
__global__ __launch_bounds__(256) void flash_attn(const __bf16* __restrict__ Q,
                                                  const __bf16* __restrict__ Kb,
                                                  const __bf16* __restrict__ Vt,
                                                  __bf16* __restrict__ O) {
    __shared__ __align__(16) __bf16 Ks[2][64][72];
    __shared__ __align__(16) __bf16 Vs[2][64][72];
    __shared__ __align__(16) __bf16 Ps[8][16][72];
    const int tid = threadIdx.x, lane = tid & 31, wid = tid >> 5;
    const int lhi = lane >> 4, llo = lane & 15;
    const int bh = blockIdx.y, b = bh >> 4, h = bh & 15;
    const int q0 = blockIdx.x * 128 + wid * 16;     // 16 query rows per wave
    const int hc = h << 6;
    const int srow = tid >> 3, scol = (tid & 7) << 3;   // staging chunk coords

    // Q fragments (A layout, 16x64 over d) live in registers for the whole kernel
    Frag qf[2];
    {
        const __bf16* qrow = Q + ((size_t)(b * 2048 + q0 + llo)) * 1024 + hc;
#pragma unroll
        for (int kf = 0; kf < 2; kf++) {
            qf[kf].h[0] = *(const v8bf*)&qrow[kf * 32 + 8 * lhi];
            qf[kf].h[1] = *(const v8bf*)&qrow[kf * 32 + 16 + 8 * lhi];
        }
    }
    float mi[8], li[8];
#pragma unroll
    for (int r = 0; r < 8; r++) { mi[r] = -1e30f; li[r] = 0.f; }
    v8f oacc[4] = {};

    const __bf16* Kbase = Kb + (size_t)(b * 2048) * 1024 + hc;
    const __bf16* Vbase = Vt + ((size_t)(b * 1024 + hc)) * 2048;

    auto stage = [&](int t0, int buf) {
#pragma unroll
        for (int half = 0; half < 2; half++) {
            int row = srow + half * 32;
            async_cp16(&Kbase[(size_t)(t0 + row) * 1024 + scol], &Ks[buf][row][scol]);
            async_cp16(&Vbase[(size_t)row * 2048 + t0 + scol],   &Vs[buf][row][scol]);
        }
    };

    stage(0, 0);
    for (int tt = 0; tt < 32; tt++) {
        async_wait0();
        __syncthreads();
        if (tt + 1 < 32) stage((tt + 1) << 6, (tt + 1) & 1);
        const int buf = tt & 1;

        // scores S = Q * K^T  (16 x 64 per wave)
        v8f sc[4] = {};
#pragma unroll
        for (int kf = 0; kf < 2; kf++) {
#pragma unroll
            for (int nt = 0; nt < 4; nt++) {
                Frag kb;
                int t = nt * 16 + llo;
                kb.h[0] = *(const v8bf*)&Ks[buf][t][kf * 32 + 8 * lhi];
                kb.h[1] = *(const v8bf*)&Ks[buf][t][kf * 32 + 16 + 8 * lhi];
                sc[nt] = wmma_bf16(qf[kf], kb, sc[nt]);
            }
        }

        // online softmax update (row stats reduced across the 16-lane half)
        float cf[8];
#pragma unroll
        for (int r = 0; r < 8; r++) {
            float mx = sc[0][r];
#pragma unroll
            for (int nt = 1; nt < 4; nt++) mx = fmaxf(mx, sc[nt][r]);
#pragma unroll
            for (int d = 1; d < 16; d <<= 1) mx = fmaxf(mx, __shfl_xor(mx, d, 32));
            float mnew = fmaxf(mi[r], mx);
            cf[r] = __expf(mi[r] - mnew);
            mi[r] = mnew;
            float rs = 0.f;
#pragma unroll
            for (int nt = 0; nt < 4; nt++) {
                float p = __expf(sc[nt][r] - mnew);
                sc[nt][r] = p;
                rs += p;
            }
#pragma unroll
            for (int d = 1; d < 16; d <<= 1) rs += __shfl_xor(rs, d, 32);
            li[r] = li[r] * cf[r] + rs;
        }

        // relayout P (C-layout) -> A-layout through per-wave LDS scratch
#pragma unroll
        for (int nt = 0; nt < 4; nt++)
#pragma unroll
            for (int r = 0; r < 8; r++)
                Ps[wid][r + 8 * lhi][nt * 16 + llo] = (__bf16)sc[nt][r];

        // rescale running O accumulator
#pragma unroll
        for (int dt = 0; dt < 4; dt++)
#pragma unroll
            for (int r = 0; r < 8; r++)
                oacc[dt][r] *= cf[r];

        // O += P * V
#pragma unroll
        for (int kf = 0; kf < 2; kf++) {
            Frag pf;
            pf.h[0] = *(const v8bf*)&Ps[wid][llo][kf * 32 + 8 * lhi];
            pf.h[1] = *(const v8bf*)&Ps[wid][llo][kf * 32 + 16 + 8 * lhi];
#pragma unroll
            for (int dt = 0; dt < 4; dt++) {
                Frag vb;
                int d = dt * 16 + llo;
                vb.h[0] = *(const v8bf*)&Vs[buf][d][kf * 32 + 8 * lhi];
                vb.h[1] = *(const v8bf*)&Vs[buf][d][kf * 32 + 16 + 8 * lhi];
                oacc[dt] = wmma_bf16(pf, vb, oacc[dt]);
            }
        }
    }

    // normalize and write out (concat-head row-major bf16)
#pragma unroll
    for (int r = 0; r < 8; r++) {
        float inv = 1.0f / li[r];
        size_t row = (size_t)(b * 2048 + q0 + r + 8 * lhi);
#pragma unroll
        for (int dt = 0; dt < 4; dt++)
            O[row * 1024 + hc + dt * 16 + llo] = (__bf16)(oacc[dt][r] * inv);
    }
}

// ---------------- host orchestration ----------------
extern "C" void kernel_launch(void* const* d_in, const int* in_sizes, int n_in,
                              void* d_out, int out_size, void* d_ws, size_t ws_size,
                              hipStream_t stream) {
    const float* x   = (const float*)d_in[0];
    const float* Wq  = (const float*)d_in[1];
    const float* bq  = (const float*)d_in[2];
    const float* Wk  = (const float*)d_in[3];
    const float* bk  = (const float*)d_in[4];
    const float* Wv  = (const float*)d_in[5];
    const float* bv  = (const float*)d_in[6];
    const float* Wo  = (const float*)d_in[7];
    const float* bo  = (const float*)d_in[8];
    const float* W1  = (const float*)d_in[9];
    const float* b1  = (const float*)d_in[10];
    const float* W2  = (const float*)d_in[11];
    const float* b2  = (const float*)d_in[12];
    const float* g1  = (const float*)d_in[13];
    const float* be1 = (const float*)d_in[14];
    const float* g2  = (const float*)d_in[15];
    const float* be2 = (const float*)d_in[16];
    float* out = (float*)d_out;

    char* ws = (char*)d_ws;
    size_t off = 0;
    auto alloc = [&](size_t bytes) -> void* {
        size_t a = (off + 255) & ~(size_t)255;
        off = a + bytes;
        return (void*)(ws + a);
    };
    const size_t R = 8192;  // B*S rows
    __bf16* hbf  = (__bf16*)alloc(R * 1024 * 2);
    __bf16* wqT  = (__bf16*)alloc(1024 * 1024 * 2);
    __bf16* wkT  = (__bf16*)alloc(1024 * 1024 * 2);
    __bf16* wvT  = (__bf16*)alloc(1024 * 1024 * 2);
    __bf16* woT  = (__bf16*)alloc(1024 * 1024 * 2);
    __bf16* w1T  = (__bf16*)alloc((size_t)4096 * 1024 * 2);
    __bf16* w2T  = (__bf16*)alloc((size_t)4096 * 1024 * 2);
    __bf16* Qbf  = (__bf16*)alloc(R * 1024 * 2);
    __bf16* Kbf  = (__bf16*)alloc(R * 1024 * 2);
    __bf16* VtB  = (__bf16*)alloc(R * 1024 * 2);
    __bf16* aO   = (__bf16*)alloc(R * 1024 * 2);
    float*  x1   = (float*) alloc(R * 1024 * 4);
    __bf16* h2bf = (__bf16*)alloc(R * 1024 * 2);
    __bf16* ff1  = (__bf16*)alloc(R * 4096 * 2);

    // weight conversion / pre-transpose to bf16
    qkv_cvt<<<4096, 256, 0, stream>>>(Wq, wqT);
    qkv_cvt<<<4096, 256, 0, stream>>>(Wk, wkT);
    qkv_cvt<<<4096, 256, 0, stream>>>(Wv, wvT);
    transpose_cvt<<<4096,  256, 0, stream>>>(Wo, woT, 1024, 1024);
    transpose_cvt<<<16384, 256, 0, stream>>>(W1, w1T, 1024, 4096);
    transpose_cvt<<<16384, 256, 0, stream>>>(W2, w2T, 4096, 1024);

    // LN1
    ln_bf16<<<8192, 256, 0, stream>>>(x, g1, be1, hbf);

    // QKV projections (scale 1/sqrt(64) folded into Q; V stored transposed)
    gemm_bf16<<<dim3(8, 64), 256, 0, stream>>>(hbf, wqT, bq, nullptr, 0.125f,
                                               8192, 1024, 1024, 0, Qbf, nullptr);
    gemm_bf16<<<dim3(8, 64), 256, 0, stream>>>(hbf, wkT, bk, nullptr, 1.0f,
                                               8192, 1024, 1024, 0, Kbf, nullptr);
    gemm_bf16<<<dim3(8, 64), 256, 0, stream>>>(hbf, wvT, bv, nullptr, 1.0f,
                                               8192, 1024, 1024, 1, VtB, nullptr);

    // flash attention: 16 q-chunks x (B*H = 64)
    flash_attn<<<dim3(16, 64), 256, 0, stream>>>(Qbf, Kbf, VtB, aO);

    // output projection + residual -> x1 (f32)
    gemm_bf16<<<dim3(8, 64), 256, 0, stream>>>(aO, woT, bo, x, 1.0f,
                                               8192, 1024, 1024, 2, nullptr, x1);
    // LN2
    ln_bf16<<<8192, 256, 0, stream>>>(x1, g2, be2, h2bf);

    // FFN: ff1 = h2@W1+b1 ; out = gelu(ff1@W2+b2) + x1
    gemm_bf16<<<dim3(32, 64), 256, 0, stream>>>(h2bf, w1T, b1, nullptr, 1.0f,
                                                8192, 4096, 1024, 0, ff1, nullptr);
    gemm_bf16<<<dim3(8, 64), 256, 0, stream>>>(ff1, w2T, b2, x1, 1.0f,
                                               8192, 1024, 4096, 3, nullptr, out);
}
